// model_41832981463624
// MI455X (gfx1250) — matrix operand order repaired and checked by reading, not compiled, not run on hardware
//
#include <hip/hip_runtime.h>
#include <hip/hip_bf16.h>

typedef __attribute__((ext_vector_type(16))) _Float16 v16h;
typedef __attribute__((ext_vector_type(8)))  _Float16 v8h;
typedef __attribute__((ext_vector_type(4)))  _Float16 v4h;
typedef __attribute__((ext_vector_type(8)))  float    v8f;

#define HID 64

// ---------------- utility kernels ----------------

__global__ void zero_f32_kernel(float* __restrict__ p, int count) {
    for (int i = blockIdx.x * blockDim.x + threadIdx.x; i < count;
         i += gridDim.x * blockDim.x)
        p[i] = 0.0f;
}

__global__ void cvt_weights_kernel(const float* __restrict__ wrel,
                                   const float* __restrict__ wroot,
                                   _Float16* __restrict__ wrel16,
                                   _Float16* __restrict__ wroot16, int count) {
    int i = blockIdx.x * blockDim.x + threadIdx.x;
    if (i < count) {
        wrel16[i]  = (_Float16)wrel[i];
        wroot16[i] = (_Float16)wroot[i];
    }
}

// ---------------- layer 0 (C_in = 1) ----------------

__global__ void scatter0_kernel(const int* __restrict__ ei,
                                const float* __restrict__ ew,
                                const float* __restrict__ x,
                                float* __restrict__ agg0, int nedges) {
    int e = blockIdx.x * blockDim.x + threadIdx.x;
    if (e >= nedges) return;
    int src = ei[e];
    int dst = ei[nedges + e];
    atomicAdd(agg0 + dst, x[src] * ew[e]);
}

__global__ void expand0_kernel(const float* __restrict__ agg0,
                               const float* __restrict__ x,
                               const float* __restrict__ Wrel0,
                               const float* __restrict__ brel0,
                               const float* __restrict__ Wroot0,
                               _Float16* __restrict__ h16, int nnodes) {
    int i = blockIdx.x * blockDim.x + threadIdx.x;
    if (i >= nnodes * HID) return;
    int n = i >> 6;
    int c = i & 63;
    float v = agg0[n] * Wrel0[c] + brel0[c] + x[n] * Wroot0[c];
    h16[i] = (_Float16)(v > 0.0f ? v : 0.0f);
}

// ---------------- edge scatter for 64-channel layers ----------------

// 16 threads per edge, 4 channels each: agg32[dst][c] += h16[src][c] * w
__global__ void scatter_mid_kernel(const int* __restrict__ ei,
                                   const float* __restrict__ ew,
                                   const _Float16* __restrict__ h16,
                                   float* __restrict__ agg32, int nedges) {
    int gid = blockIdx.x * blockDim.x + threadIdx.x;
    int e = gid >> 4;
    if (e >= nedges) return;
    int cb = (gid & 15) << 2;
    int src = ei[e];
    int dst = ei[nedges + e];
    float w = ew[e];
    v4h hv = *(const v4h*)(h16 + (size_t)src * HID + cb);
    float* ap = agg32 + (size_t)dst * HID + cb;
    atomicAdd(ap + 0, (float)hv[0] * w);
    atomicAdd(ap + 1, (float)hv[1] * w);
    atomicAdd(ap + 2, (float)hv[2] * w);
    atomicAdd(ap + 3, (float)hv[3] * w);
}

// ---------------- WMMA GEMM: hout = relu(agg @ WrelT + brel + h @ WrootT) ----------------
// 256 threads = 8 waves; each wave computes a 16(node) x 16(out-chan) tile.
// Weights (2 x 64x64 f16 = 16 KB) staged into LDS via async global->LDS DMA.
// agg is read in f32 and converted to f16 in registers (no separate cvt pass).

__global__ void gemm_relu_kernel(const float* __restrict__ agg32,
                                 const _Float16* __restrict__ h16,
                                 const _Float16* __restrict__ wrel16,
                                 const _Float16* __restrict__ wroot16,
                                 const float* __restrict__ brel,
                                 _Float16* __restrict__ hout, int nnodes) {
    __shared__ _Float16 lds_w[2 * HID * HID];   // [0..4095]=Wrel, [4096..8191]=Wroot

    const int t = threadIdx.x;
    // ---- async copy 16 KB of weights into LDS (256 lanes x 16B x 4 issues) ----
    {
        unsigned l0 = (unsigned)(unsigned long long)&lds_w[0] + (unsigned)(t * 16);
        unsigned long long grel  = (unsigned long long)wrel16  + (unsigned long long)(t * 16);
        unsigned long long groot = (unsigned long long)wroot16 + (unsigned long long)(t * 16);
        asm volatile("global_load_async_to_lds_b128 %0, %1, off"
                     :: "v"(l0),           "v"(grel)            : "memory");
        asm volatile("global_load_async_to_lds_b128 %0, %1, off"
                     :: "v"(l0 + 4096u),   "v"(grel + 4096ull)  : "memory");
        asm volatile("global_load_async_to_lds_b128 %0, %1, off"
                     :: "v"(l0 + 8192u),   "v"(groot)           : "memory");
        asm volatile("global_load_async_to_lds_b128 %0, %1, off"
                     :: "v"(l0 + 12288u),  "v"(groot + 4096ull) : "memory");
        asm volatile("s_wait_asynccnt 0x0" ::: "memory");
    }
    __syncthreads();

    const int wave = t >> 5;
    const int lane = t & 31;
    const int tile = blockIdx.x * 8 + wave;
    const int mtile = tile >> 2;   // node tile (16 nodes)
    const int ntile = tile & 3;    // output-channel tile (16 channels)
    if (mtile * 16 >= nnodes) return;   // wave-uniform, EXEC stays all-ones

    const int row = lane & 15;     // A: M row | B: weight row (=N col) | C/D: N col
    const int half = lane >> 4;
    const int m0 = mtile * 16;

    const float b = brel[ntile * 16 + row];
    v8f acc_rel  = {b, b, b, b, b, b, b, b};
    v8f acc_root = {0.f, 0.f, 0.f, 0.f, 0.f, 0.f, 0.f, 0.f};

    const _Float16* lds_wrel  = lds_w;
    const _Float16* lds_wroot = lds_w + HID * HID;

#pragma unroll
    for (int kc = 0; kc < 2; ++kc) {
        // ---- A (rel): agg f32 -> f16 in regs; lane row m0+row, K = kc*32+half*8 (+16)
        const int ka = kc * 32 + half * 8;
        const float* pa = agg32 + (size_t)(m0 + row) * HID + ka;
        v8f alo = *(const v8f*)pa;
        v8f ahi = *(const v8f*)(pa + 16);
        v16h a_rel;
#pragma unroll
        for (int i = 0; i < 8; ++i) {
            a_rel[i]     = (_Float16)alo[i];
            a_rel[8 + i] = (_Float16)ahi[i];
        }
        // ---- A (root): h16 is already f16
        const _Float16* ph = h16 + (size_t)(m0 + row) * HID + ka;
        v8h h_lo = *(const v8h*)ph;
        v8h h_hi = *(const v8h*)(ph + 16);
        v16h a_root = __builtin_shufflevector(h_lo, h_hi, 0, 1, 2, 3, 4, 5, 6, 7,
                                              8, 9, 10, 11, 12, 13, 14, 15);

        // ---- B fragments: B[k][n] = W[n][k]; lane col = row, K = kc*32+half*8 (+16)
        const int kb = kc * 32 + half * 8;
        const _Float16* pr = lds_wrel + (size_t)(ntile * 16 + row) * HID + kb;
        v8h br0 = *(const v8h*)pr;
        v8h br1 = *(const v8h*)(pr + 16);
        v16h b_rel = __builtin_shufflevector(br0, br1, 0, 1, 2, 3, 4, 5, 6, 7,
                                             8, 9, 10, 11, 12, 13, 14, 15);
        const _Float16* po = lds_wroot + (size_t)(ntile * 16 + row) * HID + kb;
        v8h bo0 = *(const v8h*)po;
        v8h bo1 = *(const v8h*)(po + 16);
        v16h b_root = __builtin_shufflevector(bo0, bo1, 0, 1, 2, 3, 4, 5, 6, 7,
                                              8, 9, 10, 11, 12, 13, 14, 15);

        // interleave two independent accumulator chains (hides WMMA->WMMA RAW)
        acc_rel  = __builtin_amdgcn_wmma_f32_16x16x32_f16(false, a_rel, false, b_rel,
                                                          (short)0, acc_rel, false, false);
        acc_root = __builtin_amdgcn_wmma_f32_16x16x32_f16(false, a_root, false, b_root,
                                                          (short)0, acc_root, false, false);
    }

    // D layout: lane -> N col = row; VGPR v -> M row = half*8 + v
#pragma unroll
    for (int v = 0; v < 8; ++v) {
        float val = acc_rel[v] + acc_root[v];
        val = val > 0.0f ? val : 0.0f;
        int m = m0 + half * 8 + v;
        hout[(size_t)m * HID + ntile * 16 + row] = (_Float16)val;
    }
}

// ---------------- last layer (C_out = 1): wave-per-node reduction ----------------

__global__ void final_kernel(const float* __restrict__ agg32,
                             const _Float16* __restrict__ h16,
                             const float* __restrict__ WrelL,
                             const float* __restrict__ brelL,
                             const float* __restrict__ WrootL,
                             float* __restrict__ out, int nnodes) {
    int n = blockIdx.x * (blockDim.x >> 5) + (threadIdx.x >> 5);
    int lane = threadIdx.x & 31;
    if (n >= nnodes) return;
    float s = 0.0f;
#pragma unroll
    for (int c = lane; c < HID; c += 32)
        s += agg32[(size_t)n * HID + c] * WrelL[c] +
             (float)h16[(size_t)n * HID + c] * WrootL[c];
#pragma unroll
    for (int off = 16; off; off >>= 1)
        s += __shfl_xor(s, off, 32);
    if (lane == 0) out[n] = s + brelL[0];
}

// ---------------- launcher ----------------

extern "C" void kernel_launch(void* const* d_in, const int* in_sizes, int n_in,
                              void* d_out, int out_size, void* d_ws, size_t ws_size,
                              hipStream_t stream) {
    const float* x         = (const float*)d_in[0];
    const int*   ei        = (const int*)d_in[1];
    const float* ew        = (const float*)d_in[2];
    const float* Wrel0     = (const float*)d_in[3];
    const float* brel0     = (const float*)d_in[4];
    const float* Wroot0    = (const float*)d_in[5];
    const float* Wrel_mid  = (const float*)d_in[6];
    const float* brel_mid  = (const float*)d_in[7];
    const float* Wroot_mid = (const float*)d_in[8];
    const float* WrelL     = (const float*)d_in[9];
    const float* brelL     = (const float*)d_in[10];
    const float* WrootL    = (const float*)d_in[11];

    const int N = in_sizes[0];      // 100000
    const int E = in_sizes[2];      // 1600000
    const int NMID = in_sizes[6] / (HID * HID);   // 3

    char* ws = (char*)d_ws;
    size_t off = 0;
    auto alloc = [&](size_t bytes) -> void* {
        void* p = ws + off;
        off = (off + bytes + 255) & ~(size_t)255;
        return p;
    };
    _Float16* hA      = (_Float16*)alloc((size_t)N * HID * sizeof(_Float16));
    _Float16* hB      = (_Float16*)alloc((size_t)N * HID * sizeof(_Float16));
    float*    agg32   = (float*)alloc((size_t)N * HID * sizeof(float));
    float*    agg0    = (float*)alloc((size_t)N * sizeof(float));
    _Float16* w16rel  = (_Float16*)alloc((size_t)NMID * HID * HID * sizeof(_Float16));
    _Float16* w16root = (_Float16*)alloc((size_t)NMID * HID * HID * sizeof(_Float16));
    (void)ws_size;

    const int B256 = 256;
    const int wcount = NMID * HID * HID;

    cvt_weights_kernel<<<(wcount + B256 - 1) / B256, B256, 0, stream>>>(
        Wrel_mid, Wroot_mid, w16rel, w16root, wcount);

    // ---- layer 0 ----
    zero_f32_kernel<<<1024, B256, 0, stream>>>(agg0, N);
    scatter0_kernel<<<(E + B256 - 1) / B256, B256, 0, stream>>>(ei, ew, x, agg0, E);
    expand0_kernel<<<(N * HID + B256 - 1) / B256, B256, 0, stream>>>(
        agg0, x, Wrel0, brel0, Wroot0, hA, N);

    // ---- middle layers ----
    _Float16* hcur = hA;
    _Float16* hnext = hB;
    const int tiles = ((N + 15) / 16) * 4;          // 16-node x 16-chan tiles
    const int gemm_blocks = (tiles + 7) / 8;        // 8 waves per block
    const long sc_threads = (long)E * 16;
    const int sc_blocks = (int)((sc_threads + B256 - 1) / B256);

    for (int i = 0; i < NMID; ++i) {
        zero_f32_kernel<<<2048, B256, 0, stream>>>(agg32, N * HID);
        scatter_mid_kernel<<<sc_blocks, B256, 0, stream>>>(ei, ew, hcur, agg32, E);
        gemm_relu_kernel<<<gemm_blocks, B256, 0, stream>>>(
            agg32, hcur, w16rel + (size_t)i * HID * HID,
            w16root + (size_t)i * HID * HID, brel_mid + (size_t)i * HID,
            hnext, N);
        _Float16* tmp = hcur; hcur = hnext; hnext = tmp;
    }

    // ---- last layer ----
    zero_f32_kernel<<<2048, B256, 0, stream>>>(agg32, N * HID);
    scatter_mid_kernel<<<sc_blocks, B256, 0, stream>>>(ei, ew, hcur, agg32, E);
    final_kernel<<<(N * 32 + B256 - 1) / B256, B256, 0, stream>>>(
        agg32, hcur, WrelL, brelL, WrootL, (float*)d_out, N);
}
